// MultiHeadAttention_56109452755642
// MI455X (gfx1250) — matrix operand order repaired and checked by reading, not compiled, not run on hardware
//
#include <hip/hip_runtime.h>
#include <hip/hip_bf16.h>

// ---------------------------------------------------------------------------
// Problem constants (from reference): B=4, S=2048, D=2048, H=16, HD=128
// ---------------------------------------------------------------------------
#define BB 4
#define SS 2048
#define DD 2048
#define HH 16
#define HDD 128
#define MM (BB * SS)        // 8192 rows in all projection GEMMs
#define KK DD               // 2048 reduction dim

typedef __attribute__((ext_vector_type(16))) __bf16 bf16x16;
typedef __attribute__((ext_vector_type(8)))  float  f32x8;

union Frag {                // 32 bytes = one WMMA 16-bit A/B operand per lane
    uint4   q[2];
    bf16x16 v;
};

__device__ __forceinline__ unsigned short f2bf(float f) {
    unsigned u = __float_as_uint(f);
    unsigned r = u + 0x7fffu + ((u >> 16) & 1u);   // round-to-nearest-even
    return (unsigned short)(r >> 16);
}

__device__ __forceinline__ f32x8 wmma_bf16(bf16x16 a, bf16x16 b, f32x8 c) {
    return __builtin_amdgcn_wmma_f32_16x16x32_bf16(
        /*neg_a=*/false, a, /*neg_b=*/false, b,
        /*c_mod=*/(short)0, c, /*reuse_a=*/false, /*reuse_b=*/false);
}

// ---------------------------------------------------------------------------
// f32 -> bf16 cast kernel
// ---------------------------------------------------------------------------
__global__ __launch_bounds__(256)
void cast_bf16_kernel(const float* __restrict__ in,
                      unsigned short* __restrict__ out, int n) {
    int i = blockIdx.x * 256 + threadIdx.x;
    if (i < n) out[i] = f2bf(in[i]);
}

// ---------------------------------------------------------------------------
// NT GEMM: Y[m,n] = sum_k X[m,k]*W[n,k] + bias[n]
// One wave -> 64x64 tile (4x4 WMMA accumulators), block = 8 waves, K step 32.
// The 64x32 A tile is shared by all 8 waves: staged into LDS once per block
// via async global->LDS copies (ASYNCcnt), double-buffered across K-steps.
// MODE 0: scatter bf16 into [B,H,S,HD]    (Q, K)
// MODE 1: scatter bf16 into [B,H,HD,S]    (V transposed)
// MODE 2: f32 [M,N]                       (final projection -> d_out)
// ---------------------------------------------------------------------------
template<int MODE>
__global__ __launch_bounds__(256)
void gemm_bf16_nt(const unsigned short* __restrict__ X,
                  const unsigned short* __restrict__ W,
                  const float* __restrict__ bias,
                  void* __restrict__ out) {
    __shared__ unsigned short atile[2][64 * 32];   // double-buffered A tile

    const int tid  = threadIdx.x;
    const int wave = tid >> 5;
    const int lane = tid & 31;
    const int half = lane >> 4;
    const int l16  = lane & 15;
    const int tileM = blockIdx.x * 64;
    const int tileN = (blockIdx.y * 8 + wave) * 64;

    // Each of the 256 threads copies 16B of the 64x32 A tile (4KB total).
    const int arow = tid >> 2;          // 0..63
    const int acol = (tid & 3) * 8;     // 0,8,16,24

    auto issueA = [&](int kk, int buf) {
        const unsigned short* gp = X + (size_t)(tileM + arow) * KK + kk + acol;
        // Flat LDS address: offset lives in addr[31:0] (ISA aperture rule).
        unsigned lds = (unsigned)(uintptr_t)&atile[buf][arow * 32 + acol];
        unsigned long long ga = (unsigned long long)(uintptr_t)gp;
        asm volatile("global_load_async_to_lds_b128 %0, %1, off"
                     :: "v"(lds), "v"(ga) : "memory");
    };

    f32x8 acc[4][4] = {};

    issueA(0, 0);
    int cur = 0;
    for (int kk = 0; kk < KK; kk += 32) {
        if (kk + 32 < KK) {
            issueA(kk + 32, cur ^ 1);                       // prefetch next
            asm volatile("s_wait_asynccnt 0x1" ::: "memory"); // current done
        } else {
            asm volatile("s_wait_asynccnt 0x0" ::: "memory");
        }
        __syncthreads();   // all waves' slices of atile[cur] are resident

        Frag a[4], b[4];
#pragma unroll
        for (int i = 0; i < 4; ++i) {
            // A layout: elems 0..7 -> K = half*8.., elems 8..15 -> +16
            const unsigned short* p =
                &atile[cur][(i * 16 + l16) * 32 + half * 8];
            a[i].q[0] = *(const uint4*)(p);
            a[i].q[1] = *(const uint4*)(p + 16);
        }
#pragma unroll
        for (int j = 0; j < 4; ++j) {
            // B layout: elems 0..7 -> K = half*8.., elems 8..15 -> +16
            const unsigned short* p =
                W + (size_t)(tileN + j * 16 + l16) * KK + kk + half * 8;
            b[j].q[0] = *(const uint4*)(p);
            b[j].q[1] = *(const uint4*)(p + 16);
        }
#pragma unroll
        for (int i = 0; i < 4; ++i)
#pragma unroll
            for (int j = 0; j < 4; ++j)
                acc[i][j] = wmma_bf16(a[i].v, b[j].v, acc[i][j]);

        __syncthreads();   // everyone done reading atile[cur] before overwrite
        cur ^= 1;
    }

    // Epilogue. C layout: VGPR r -> row (r + 8*half), lane l16 -> col.
#pragma unroll
    for (int i = 0; i < 4; ++i) {
#pragma unroll
        for (int j = 0; j < 4; ++j) {
            const int n  = tileN + j * 16 + l16;
            const float bv = bias[n];
#pragma unroll
            for (int r = 0; r < 8; ++r) {
                const int   m = tileM + i * 16 + r + half * 8;
                const float v = acc[i][j][r] + bv;
                if constexpr (MODE == 2) {
                    ((float*)out)[(size_t)m * DD + n] = v;
                } else {
                    const int bb = m / SS, s = m % SS;
                    const int h = n / HDD, hd = n % HDD;
                    size_t idx;
                    if constexpr (MODE == 0)
                        idx = (((size_t)bb * HH + h) * SS + s) * HDD + hd;
                    else
                        idx = (((size_t)bb * HH + h) * HDD + hd) * SS + s;
                    ((unsigned short*)out)[idx] = f2bf(v);
                }
            }
        }
    }
}

// ---------------------------------------------------------------------------
// Flash attention. One wave owns 16 query rows of one (b,h); block = 8 waves.
// Streams keys/values in 32-column chunks; online softmax in registers;
// P re-laid out C-format -> A-format through LDS (1 KB per wave).
//   Q : [B,H,S,HD] bf16     Kh : [B,H,S,HD] bf16     Vt : [B,H,HD,S] bf16
//   Out: [B,S,D] bf16 (heads re-interleaved for the final projection)
// ---------------------------------------------------------------------------
__global__ __launch_bounds__(256)
void flash_attn_kernel(const unsigned short* __restrict__ Q,
                       const unsigned short* __restrict__ Kh,
                       const unsigned short* __restrict__ Vt,
                       unsigned short* __restrict__ Out) {
    __shared__ unsigned short plds[8][16 * 32];

    const int tid  = threadIdx.x;
    const int wave = tid >> 5;
    const int lane = tid & 31;
    const int half = lane >> 4;
    const int l16  = lane & 15;
    const int bh    = blockIdx.x;             // 0 .. B*H-1
    const int qBase = blockIdx.y * 128 + wave * 16;

    const unsigned short* qp = Q  + (size_t)bh * SS * HDD;
    const unsigned short* kp = Kh + (size_t)bh * SS * HDD;
    const unsigned short* vp = Vt + (size_t)bh * HDD * SS;
    const float scale = 0.0883883476483184f;  // 1/sqrt(128)

    // Preload Q fragments: 4 K-chunks of 32 along HD, A-format.
    Frag qf[4];
#pragma unroll
    for (int c = 0; c < 4; ++c) {
        const unsigned short* p =
            qp + (size_t)(qBase + l16) * HDD + c * 32 + half * 8;
        qf[c].q[0] = *(const uint4*)(p);
        qf[c].q[1] = *(const uint4*)(p + 16);
    }

    f32x8 o[8] = {};                          // 16 x 128 f32 output accumulator
    float m_run[8], l_run[8];
#pragma unroll
    for (int r = 0; r < 8; ++r) { m_run[r] = -1e30f; l_run[r] = 0.0f; }

    for (int kc = 0; kc < SS; kc += 32) {
        // ---- scores S = (Q K^T) * scale : two 16x16 tiles -----------------
        f32x8 s[2] = {};
#pragma unroll
        for (int j = 0; j < 2; ++j) {
#pragma unroll
            for (int c = 0; c < 4; ++c) {
                Frag kf;
                const unsigned short* p =
                    kp + (size_t)(kc + j * 16 + l16) * HDD + c * 32 + half * 8;
                kf.q[0] = *(const uint4*)(p);
                kf.q[1] = *(const uint4*)(p + 16);
                s[j] = wmma_bf16(qf[c].v, kf.v, s[j]);
            }
        }
#pragma unroll
        for (int j = 0; j < 2; ++j)
#pragma unroll
            for (int r = 0; r < 8; ++r) s[j][r] *= scale;

        // ---- online softmax (row r lives in VGPR r of each 16-lane half) --
        float alpha[8];
#pragma unroll
        for (int r = 0; r < 8; ++r) {
            float mx = fmaxf(s[0][r], s[1][r]);
#pragma unroll
            for (int d = 1; d < 16; d <<= 1)
                mx = fmaxf(mx, __shfl_xor(mx, d, 32));
            const float mn = fmaxf(m_run[r], mx);
            alpha[r] = __expf(m_run[r] - mn);
            m_run[r] = mn;
            const float p0 = __expf(s[0][r] - mn);
            const float p1 = __expf(s[1][r] - mn);
            s[0][r] = p0; s[1][r] = p1;
            float rs = p0 + p1;
#pragma unroll
            for (int d = 1; d < 16; d <<= 1)
                rs += __shfl_xor(rs, d, 32);
            l_run[r] = l_run[r] * alpha[r] + rs;
        }
#pragma unroll
        for (int t = 0; t < 8; ++t)
#pragma unroll
            for (int r = 0; r < 8; ++r) o[t][r] *= alpha[r];

        // ---- P: C-format -> A-format via LDS ------------------------------
#pragma unroll
        for (int j = 0; j < 2; ++j)
#pragma unroll
            for (int r = 0; r < 8; ++r)
                plds[wave][(r + half * 8) * 32 + j * 16 + l16] = f2bf(s[j][r]);
        __syncthreads();
        Frag pf;
        {
            const unsigned short* p = &plds[wave][l16 * 32 + half * 8];
            pf.q[0] = *(const uint4*)(p);
            pf.q[1] = *(const uint4*)(p + 16);
        }
        __syncthreads();

        // ---- O += P @ V (V transposed: columns of V contiguous) ----------
#pragma unroll
        for (int t = 0; t < 8; ++t) {
            Frag vf;
            const unsigned short* p =
                vp + (size_t)(t * 16 + l16) * SS + kc + half * 8;
            vf.q[0] = *(const uint4*)(p);
            vf.q[1] = *(const uint4*)(p + 16);
            o[t] = wmma_bf16(pf.v, vf.v, o[t]);
        }
    }

    // ---- finalize: divide by row sums, scatter heads back to [B,S,D] ------
    const int b = bh / HH, h = bh % HH;
#pragma unroll
    for (int r = 0; r < 8; ++r) {
        const float inv = 1.0f / l_run[r];
        const int srow = qBase + r + half * 8;
#pragma unroll
        for (int t = 0; t < 8; ++t) {
            const int dcol = h * HDD + t * 16 + l16;
            Out[((size_t)b * SS + srow) * DD + dcol] = f2bf(o[t][r] * inv);
        }
    }
}

// ---------------------------------------------------------------------------
// Launcher
// ---------------------------------------------------------------------------
extern "C" void kernel_launch(void* const* d_in, const int* in_sizes, int n_in,
                              void* d_out, int out_size, void* d_ws, size_t ws_size,
                              hipStream_t stream) {
    const float* x  = (const float*)d_in[0];
    const float* Wq = (const float*)d_in[1];
    const float* bq = (const float*)d_in[2];
    const float* Wk = (const float*)d_in[3];
    const float* bk = (const float*)d_in[4];
    const float* Wv = (const float*)d_in[5];
    const float* bv = (const float*)d_in[6];
    const float* Wo = (const float*)d_in[7];
    const float* bo = (const float*)d_in[8];

    const size_t XE = (size_t)MM * DD;   // 16,777,216 elements
    const size_t WE = (size_t)DD * DD;   //  4,194,304 elements

    unsigned short* xb   = (unsigned short*)d_ws;
    unsigned short* wqb  = xb  + XE;
    unsigned short* wkb  = wqb + WE;
    unsigned short* wvb  = wkb + WE;
    unsigned short* wob  = wvb + WE;
    unsigned short* qh   = wob + WE;     // [B,H,S,HD]
    unsigned short* kh   = qh  + XE;     // [B,H,S,HD]
    unsigned short* vt   = kh  + XE;     // [B,H,HD,S]
    unsigned short* attn = vt  + XE;     // [B,S,D]

    // 1) cast to bf16
    cast_bf16_kernel<<<dim3((unsigned)(XE / 256)), 256, 0, stream>>>(x,  xb,  (int)XE);
    cast_bf16_kernel<<<dim3((unsigned)(WE / 256)), 256, 0, stream>>>(Wq, wqb, (int)WE);
    cast_bf16_kernel<<<dim3((unsigned)(WE / 256)), 256, 0, stream>>>(Wk, wkb, (int)WE);
    cast_bf16_kernel<<<dim3((unsigned)(WE / 256)), 256, 0, stream>>>(Wv, wvb, (int)WE);
    cast_bf16_kernel<<<dim3((unsigned)(WE / 256)), 256, 0, stream>>>(Wo, wob, (int)WE);

    // 2) QKV projections (64x64 per wave; grid covers 8192 x 2048)
    dim3 gg(MM / 64, DD / (64 * 8)), bb(256);
    gemm_bf16_nt<0><<<gg, bb, 0, stream>>>(xb, wqb, bq, qh);
    gemm_bf16_nt<0><<<gg, bb, 0, stream>>>(xb, wkb, bk, kh);
    gemm_bf16_nt<1><<<gg, bb, 0, stream>>>(xb, wvb, bv, vt);

    // 3) flash attention: grid = (B*H, S / (16 rows * 8 waves))
    flash_attn_kernel<<<dim3(BB * HH, SS / 128), 256, 0, stream>>>(qh, kh, vt, attn);

    // 4) output projection -> f32 d_out
    gemm_bf16_nt<2><<<gg, bb, 0, stream>>>(attn, wob, bo, (float*)d_out);
}
